// Multihead_Attention_47347719471748
// MI455X (gfx1250) — matrix operand order repaired and checked by reading, not compiled, not run on hardware
//
#include <hip/hip_runtime.h>
#include <hip/hip_bf16.h>

// Problem constants (reference: N=2, Tq=Tk=2048, C=HIDDEN=1024, 16 heads x 64)
#define NB 2
#define TQ 2048
#define TK 2048
#define CD 1024
#define NHEADS 16
#define DH 64
#define NEGVAL (-4.2949673e9f)   // -2^32 + 1

typedef __bf16 bf16_t;
typedef __attribute__((ext_vector_type(16))) __bf16 v16bf;
typedef __attribute__((ext_vector_type(8)))  __bf16 v8bf;
typedef __attribute__((ext_vector_type(8)))  float  v8f;
typedef __attribute__((ext_vector_type(4)))  float  v4f;
typedef __attribute__((ext_vector_type(4)))  unsigned uint4v;
typedef __attribute__((ext_vector_type(8)))  int   int8v;
typedef __attribute__((ext_vector_type(4)))  int   int4v;

// Tensor Data Mover availability (compile-safe: falls back to register staging)
#if defined(__has_builtin)
#  if __has_builtin(__builtin_amdgcn_tensor_load_to_lds) && \
      __has_builtin(__builtin_amdgcn_s_wait_tensorcnt)
#    define USE_TDM 1
#  endif
#endif
#ifndef USE_TDM
#  define USE_TDM 0
#endif

union Frag16 { v16bf v; v8bf h[2]; };

__device__ __forceinline__ unsigned short f2bfbits(float f) {
  unsigned u = __builtin_bit_cast(unsigned, f);
  unsigned r = u + 0x7FFFu + ((u >> 16) & 1u);   // round-to-nearest-even
  return (unsigned short)(r >> 16);
}
__device__ __forceinline__ bf16_t f2bf(float f) {
  unsigned short s = f2bfbits(f);
  return __builtin_bit_cast(bf16_t, s);
}
__device__ __forceinline__ v8bf pack8(v4f a, v4f b) {
  v8bf r;
  r[0] = f2bf(a[0]); r[1] = f2bf(a[1]); r[2] = f2bf(a[2]); r[3] = f2bf(a[3]);
  r[4] = f2bf(b[0]); r[5] = f2bf(b[1]); r[6] = f2bf(b[2]); r[7] = f2bf(b[3]);
  return r;
}
__device__ __forceinline__ v8f wmma_bf16(v16bf a, v16bf b, v8f c) {
  return __builtin_amdgcn_wmma_f32_16x16x32_bf16(false, a, false, b, (short)0, c,
                                                 false, false);
}
__device__ __forceinline__ v16bf ld_frag(const bf16_t* p0, const bf16_t* p1) {
  Frag16 f;
  f.h[0] = *(const v8bf*)p0;
  f.h[1] = *(const v8bf*)p1;
  return f.v;
}

#if USE_TDM
// Issue a TDM DMA of a 64(keys) x 64(dh) bf16 tile (row stride CD elements)
// from global memory into LDS.  D# per CDNA5 ISA ch.8.
__device__ __forceinline__ void tdm_load_tile(const bf16_t* gsrc, unsigned lds_off) {
  unsigned long long ga = (unsigned long long)(__UINTPTR_TYPE__)gsrc;
  uint4v g0;
  g0[0] = 1u;                                    // count=1, user descriptor
  g0[1] = lds_off;                               // lds_addr (bytes)
  g0[2] = (unsigned)ga;                          // global_addr[31:0]
  g0[3] = (unsigned)((ga >> 32) & 0x1FFFFFFu) | (2u << 30);  // addr[56:32] | type=2
  int8v g1;
  g1[0] = (int)(1u << 16);                       // data_size=1 (2 bytes)
  g1[1] = (int)(1024u << 16);                    // tensor_dim0 = CD (elements)
  g1[2] = (int)(4096u << 16);                    // tensor_dim1 = N*T rows
  g1[3] = (int)(64u << 16);                      // tile_dim0 = 64
  g1[4] = 64;                                    // tile_dim1 = 64
  g1[5] = 1024;                                  // tensor_dim0_stride = CD
  g1[6] = 0;
  g1[7] = 0;
  int4v z4 = {0, 0, 0, 0};
#if __clang_major__ >= 23
  int8v z8 = {0, 0, 0, 0, 0, 0, 0, 0};
  __builtin_amdgcn_tensor_load_to_lds(g0, g1, z4, z4, z8, 0);
#else
  __builtin_amdgcn_tensor_load_to_lds(g0, g1, z4, z4, 0);
#endif
}
#endif

// ---------------------------------------------------------------------------
// mask[row] = sign(sum |X[row,:]|)
// ---------------------------------------------------------------------------
__global__ __launch_bounds__(256)
void mask_kernel(const float* __restrict__ X, float* __restrict__ mask) {
  int row = blockIdx.x, tid = threadIdx.x;
  const float* x = X + (size_t)row * CD;
  float s = 0.f;
#pragma unroll
  for (int i = 0; i < 4; ++i) s += fabsf(x[tid + i * 256]);
#pragma unroll
  for (int m = 1; m < 32; m <<= 1) s += __shfl_xor(s, m, 32);
  __shared__ float red[8];
  if ((tid & 31) == 0) red[tid >> 5] = s;
  __syncthreads();
  if (tid == 0) {
    float t = 0.f;
#pragma unroll
    for (int w = 0; w < 8; ++w) t += red[w];
    mask[row] = (t > 0.f) ? 1.f : 0.f;
  }
}

// ---------------------------------------------------------------------------
// Y = relu(X @ W + b):  M=4096, K=1024, N=1024.  Block tile 128x128, 8 waves,
// wave tile 32x64, K-step 32.  Software-pipelined: next tile's global loads
// are parked in VGPRs while WMMAs run on the current ping-pong LDS buffer;
// one barrier per K-step.
// ---------------------------------------------------------------------------
__global__ __launch_bounds__(256)
void proj_kernel(const float* __restrict__ X, const float* __restrict__ W,
                 const float* __restrict__ bias, bf16_t* __restrict__ Yb,
                 float* __restrict__ Yf) {
  __shared__ __align__(16) bf16_t A_s[2][128 * 32];   // [m][k] row-major
  __shared__ __align__(16) bf16_t Wt_s[2][128 * 32];  // [n][k] (W transposed)
  int tid = threadIdx.x;
  int l = tid & 31, w = tid >> 5;
  int wvn = w & 1, wvm = w >> 1;            // wave grid: 4 (M) x 2 (N)
  int n0 = blockIdx.x * 128;
  int m0 = blockIdx.y * 128;
  int col16 = l & 15;
  int c0 = (l < 16) ? 0 : 8;                // A-frag K-chunk base (ISA layout)
  int koffB = (l < 16) ? 0 : 16;            // contiguous K half (unused; B-frag uses c0)
  int rofs = (l < 16) ? 0 : 8;              // C-layout row offset

  v8f zero = {};
  v8f acc[2][4];
#pragma unroll
  for (int i = 0; i < 2; ++i)
#pragma unroll
    for (int j = 0; j < 4; ++j) acc[i][j] = zero;

  int arow = tid >> 1;            // A loader: row 0..127
  int acolh = (tid & 1) * 16;     // A loader: col half
  int wk = tid >> 3;              // W loader: k 0..31
  int wn16 = (tid & 7) * 16;      // W loader: n chunk

  auto loadA = [&](int k0, v4f* r) {
    const float* src = X + (size_t)(m0 + arow) * CD + k0 + acolh;
#pragma unroll
    for (int q = 0; q < 4; ++q) r[q] = *(const v4f*)(src + q * 4);
  };
  auto loadW = [&](int k0, v4f* r) {
    const float* src = W + (size_t)(k0 + wk) * CD + n0 + wn16;
#pragma unroll
    for (int q = 0; q < 4; ++q) r[q] = *(const v4f*)(src + q * 4);
  };
  auto storeA = [&](int b, const v4f* r) {
    *(v8bf*)&A_s[b][arow * 32 + acolh]     = pack8(r[0], r[1]);
    *(v8bf*)&A_s[b][arow * 32 + acolh + 8] = pack8(r[2], r[3]);
  };
  auto storeW = [&](int b, const v4f* r) {
#pragma unroll
    for (int e = 0; e < 16; ++e)
      Wt_s[b][(wn16 + e) * 32 + wk] = f2bf(r[e >> 2][e & 3]);
  };

  v4f ar[4], wr[4];
  loadA(0, ar); loadW(0, wr);
  storeA(0, ar); storeW(0, wr);
  __syncthreads();

  for (int it = 0; it < CD / 32; ++it) {
    int b = it & 1;
    bool more = (it + 1) < CD / 32;
    if (more) { loadA((it + 1) * 32, ar); loadW((it + 1) * 32, wr); }

    v16bf af[2], bfrag[4];
#pragma unroll
    for (int i = 0; i < 2; ++i) {
      const bf16_t* base = &A_s[b][(wvm * 32 + i * 16 + col16) * 32];
      af[i] = ld_frag(base + c0, base + c0 + 16);
    }
#pragma unroll
    for (int j = 0; j < 4; ++j) {
      const bf16_t* base = &Wt_s[b][(wvn * 64 + j * 16 + col16) * 32];
      bfrag[j] = ld_frag(base + c0, base + c0 + 16);
    }
#pragma unroll
    for (int i = 0; i < 2; ++i)
#pragma unroll
      for (int j = 0; j < 4; ++j)
        acc[i][j] = wmma_bf16(af[i], bfrag[j], acc[i][j]);

    if (more) { storeA(1 - b, ar); storeW(1 - b, wr); }
    __syncthreads();
  }

  // epilogue: bias + relu, store bf16 (+ optional f32)
#pragma unroll
  for (int j = 0; j < 4; ++j) {
    int nl = wvn * 64 + j * 16 + col16;
    float bj = bias[n0 + nl];
#pragma unroll
    for (int i = 0; i < 2; ++i) {
#pragma unroll
      for (int r = 0; r < 8; ++r) {
        int ml = wvm * 32 + i * 16 + r + rofs;
        float v = acc[i][j][r] + bj;
        v = v > 0.f ? v : 0.f;
        size_t idx = (size_t)(m0 + ml) * CD + n0 + nl;
        Yb[idx] = f2bf(v);
        if (Yf) Yf[idx] = v;
      }
    }
  }
}

// ---------------------------------------------------------------------------
// Flash attention per (batch, head): block = 128 query rows (8 waves x 16),
// key tiles of 64, online softmax.  Pipelined ping-pong LDS staging:
// K tile via Tensor Data Mover (wave 0, TENSORcnt) when available,
// V tile via registers (needs transpose), one barrier per key tile.
// ---------------------------------------------------------------------------
__global__ __launch_bounds__(256)
void attn_kernel(const bf16_t* __restrict__ Qb, const bf16_t* __restrict__ Kb,
                 const bf16_t* __restrict__ Vb, const float* __restrict__ kmask,
                 const float* __restrict__ qmask, float* __restrict__ O) {
  __shared__ __align__(16) bf16_t K_s[2][64 * 64];     // [k][d] row-major
  __shared__ __align__(16) bf16_t Vt_s[2][64 * 64];    // [d][k] (V transposed)
  __shared__ __align__(16) bf16_t P_s[8 * 16 * 64];    // wave-private P tiles
  int tid = threadIdx.x;
  int l = tid & 31, w = tid >> 5;
  int qt = blockIdx.x & 15;           // Tq/128 tiles
  int nh = blockIdx.x >> 4;
  int h = nh & 15;
  int n = nh >> 4;
  int col16 = l & 15;
  int c0 = (l < 16) ? 0 : 8;
  int rofs = (l < 16) ? 0 : 8;
  int bhalf = (l < 16) ? 0 : 16;
  bf16_t* Pw = &P_s[w * 16 * 64];

  // chunk mapping for cooperative staging (512 chunks of 8 bf16, 2 per thread)
  int kr0 = tid >> 3, off0 = (tid & 7) * 8;
  int kr1 = (tid + 256) >> 3, off1 = ((tid + 256) & 7) * 8;

  auto loadV = [&](int kt, v8bf* vr) {
    vr[0] = *(const v8bf*)(Vb + (size_t)(n * TK + kt * 64 + kr0) * CD + h * DH + off0);
    vr[1] = *(const v8bf*)(Vb + (size_t)(n * TK + kt * 64 + kr1) * CD + h * DH + off1);
  };
  auto storeV = [&](int b, const v8bf* vr) {
#pragma unroll
    for (int e = 0; e < 8; ++e) {
      Vt_s[b][(off0 + e) * 64 + kr0] = vr[0][e];
      Vt_s[b][(off1 + e) * 64 + kr1] = vr[1][e];
    }
  };
#if !USE_TDM
  auto loadK = [&](int kt, v8bf* kr) {
    kr[0] = *(const v8bf*)(Kb + (size_t)(n * TK + kt * 64 + kr0) * CD + h * DH + off0);
    kr[1] = *(const v8bf*)(Kb + (size_t)(n * TK + kt * 64 + kr1) * CD + h * DH + off1);
  };
  auto storeK = [&](int b, const v8bf* kr) {
    *(v8bf*)&K_s[b][kr0 * 64 + off0] = kr[0];
    *(v8bf*)&K_s[b][kr1 * 64 + off1] = kr[1];
  };
#endif

  // Q fragments held in registers for the whole K loop
  int qrow = qt * 128 + w * 16 + col16;
  const bf16_t* qptr = Qb + (size_t)(n * TQ + qrow) * CD + h * DH;
  v16bf qa[2];
#pragma unroll
  for (int f = 0; f < 2; ++f)
    qa[f] = ld_frag(qptr + f * 32 + c0, qptr + f * 32 + c0 + 16);

  v8f zero = {};
  v8f oacc[4];
#pragma unroll
  for (int d = 0; d < 4; ++d) oacc[d] = zero;
  float m_run[8], l_run[8];
#pragma unroll
  for (int r = 0; r < 8; ++r) { m_run[r] = -3.0e38f; l_run[r] = 0.f; }

  // ---- prologue: stage tile 0 into buffer 0 ----
  v8bf vr[2];
#if USE_TDM
  if (w == 0)
    tdm_load_tile(Kb + (size_t)(n * TK) * CD + h * DH,
                  (unsigned)(__UINTPTR_TYPE__)&K_s[0][0]);
  loadV(0, vr);
  storeV(0, vr);
  if (w == 0) __builtin_amdgcn_s_wait_tensorcnt((short)0);
#else
  v8bf kreg[2];
  loadK(0, kreg); loadV(0, vr);
  storeK(0, kreg); storeV(0, vr);
#endif
  __syncthreads();

  for (int kt = 0; kt < TK / 64; ++kt) {
    int b = kt & 1;
    bool more = (kt + 1) < TK / 64;
    if (more) {
#if USE_TDM
      if (w == 0)
        tdm_load_tile(Kb + (size_t)(n * TK + (kt + 1) * 64) * CD + h * DH,
                      (unsigned)(__UINTPTR_TYPE__)&K_s[1 - b][0]);
#else
      loadK(kt + 1, kreg);
#endif
      loadV(kt + 1, vr);
    }

    // ---- S = Q . K^T (4 key subtiles of 16, K-dim = DH = 2 WMMA steps) ----
    v8f sacc[4];
#pragma unroll
    for (int kk = 0; kk < 4; ++kk) {
      int kidx = kk * 16 + col16;
      v8f z = zero;
#pragma unroll
      for (int f = 0; f < 2; ++f) {
        const bf16_t* base = &K_s[b][kidx * 64 + f * 32 + c0];
        v16bf kb = ld_frag(base, base + 16);
        z = wmma_bf16(qa[f], kb, z);
      }
      sacc[kk] = z;
    }

    // ---- scale + key mask ----
    float S[4][8];
#pragma unroll
    for (int kk = 0; kk < 4; ++kk) {
      float km = kmask[n * TK + kt * 64 + kk * 16 + col16];
#pragma unroll
      for (int r = 0; r < 8; ++r) {
        float s = sacc[kk][r] * 0.125f;          // 1/sqrt(64)
        S[kk][r] = (km == 0.f) ? NEGVAL : s;
      }
    }

    // ---- online softmax (row groups = 16 lanes; shfl_xor 1,2,4,8) ----
#pragma unroll
    for (int r = 0; r < 8; ++r) {
      float mloc = S[0][r];
#pragma unroll
      for (int kk = 1; kk < 4; ++kk) mloc = fmaxf(mloc, S[kk][r]);
#pragma unroll
      for (int m = 1; m < 16; m <<= 1) mloc = fmaxf(mloc, __shfl_xor(mloc, m, 32));
      float mnew = fmaxf(m_run[r], mloc);
      float alpha = __expf(m_run[r] - mnew);
      float ps = 0.f;
#pragma unroll
      for (int kk = 0; kk < 4; ++kk) {
        float p = __expf(S[kk][r] - mnew);
        S[kk][r] = p;
        ps += p;
      }
#pragma unroll
      for (int m = 1; m < 16; m <<= 1) ps += __shfl_xor(ps, m, 32);
      l_run[r] = l_run[r] * alpha + ps;
      m_run[r] = mnew;
#pragma unroll
      for (int d = 0; d < 4; ++d) oacc[d][r] *= alpha;
    }

    // ---- P (C-layout) -> wave-private LDS tile (A-layout source) ----
#pragma unroll
    for (int kk = 0; kk < 4; ++kk)
#pragma unroll
      for (int r = 0; r < 8; ++r)
        Pw[(r + rofs) * 64 + kk * 16 + col16] = f2bf(S[kk][r]);

    // ---- O += P . V (2 key chunks of 32, 4 d-subtiles of 16) ----
    v16bf pa[2];
#pragma unroll
    for (int kc = 0; kc < 2; ++kc) {
      const bf16_t* base = &Pw[col16 * 64 + kc * 32 + c0];
      pa[kc] = ld_frag(base, base + 16);
    }
#pragma unroll
    for (int d = 0; d < 4; ++d) {
#pragma unroll
      for (int kc = 0; kc < 2; ++kc) {
        const bf16_t* base = &Vt_s[b][(d * 16 + col16) * 64 + kc * 32 + c0];
        v16bf vb = ld_frag(base, base + 16);
        oacc[d] = wmma_bf16(pa[kc], vb, oacc[d]);
      }
    }

    // ---- commit next tile's staging, then the single barrier ----
    if (more) {
#if !USE_TDM
      storeK(1 - b, kreg);
#endif
      storeV(1 - b, vr);
#if USE_TDM
      if (w == 0) __builtin_amdgcn_s_wait_tensorcnt((short)0);
#endif
    }
    __syncthreads();
  }

  // ---- epilogue: 1/l, query mask, store f32 ----
#pragma unroll
  for (int r = 0; r < 8; ++r) {
    int row = qt * 128 + w * 16 + r + rofs;
    float qm = qmask[n * TQ + row];
    float inv = qm / l_run[r];
#pragma unroll
    for (int d = 0; d < 4; ++d)
      O[(size_t)(n * TQ + row) * CD + h * DH + d * 16 + col16] = oacc[d][r] * inv;
  }
}

// ---------------------------------------------------------------------------
// out = LayerNorm(O + Ql) * g + b
// ---------------------------------------------------------------------------
__global__ __launch_bounds__(256)
void ln_kernel(const float* __restrict__ O, const float* __restrict__ R,
               const float* __restrict__ g, const float* __restrict__ b,
               float* __restrict__ out) {
  int row = blockIdx.x, tid = threadIdx.x;
  const float* o = O + (size_t)row * CD;
  const float* rr = R + (size_t)row * CD;
  float x[4], s = 0.f, s2 = 0.f;
#pragma unroll
  for (int i = 0; i < 4; ++i) {
    int c = tid + i * 256;
    x[i] = o[c] + rr[c];
    s += x[i];
    s2 += x[i] * x[i];
  }
#pragma unroll
  for (int m = 1; m < 32; m <<= 1) {
    s += __shfl_xor(s, m, 32);
    s2 += __shfl_xor(s2, m, 32);
  }
  __shared__ float rs[8], rs2[8];
  if ((tid & 31) == 0) { rs[tid >> 5] = s; rs2[tid >> 5] = s2; }
  __syncthreads();
  float ts = 0.f, t2 = 0.f;
#pragma unroll
  for (int w = 0; w < 8; ++w) { ts += rs[w]; t2 += rs2[w]; }
  float mu = ts * (1.f / CD);
  float var = t2 * (1.f / CD) - mu * mu;
  float rstd = rsqrtf(var + 1e-5f);
#pragma unroll
  for (int i = 0; i < 4; ++i) {
    int c = tid + i * 256;
    out[(size_t)row * CD + c] = (x[i] - mu) * rstd * g[c] + b[c];
  }
}

// ---------------------------------------------------------------------------
extern "C" void kernel_launch(void* const* d_in, const int* in_sizes, int n_in,
                              void* d_out, int out_size, void* d_ws, size_t ws_size,
                              hipStream_t stream) {
  (void)in_sizes; (void)n_in; (void)out_size; (void)ws_size;
  const float* Q   = (const float*)d_in[0];
  const float* K   = (const float*)d_in[1];
  const float* V   = (const float*)d_in[2];
  const float* Wq  = (const float*)d_in[3];
  const float* bq  = (const float*)d_in[4];
  const float* Wk  = (const float*)d_in[5];
  const float* bk  = (const float*)d_in[6];
  const float* Wv  = (const float*)d_in[7];
  const float* bv  = (const float*)d_in[8];
  const float* lnw = (const float*)d_in[9];
  const float* lnb = (const float*)d_in[10];

  char* ws = (char*)d_ws;
  size_t off = 0;
  bf16_t* Qb = (bf16_t*)(ws + off); off += (size_t)NB * TQ * CD * 2;
  bf16_t* Kb = (bf16_t*)(ws + off); off += (size_t)NB * TK * CD * 2;
  bf16_t* Vb = (bf16_t*)(ws + off); off += (size_t)NB * TK * CD * 2;
  float* Qlf = (float*)(ws + off);  off += (size_t)NB * TQ * CD * 4;
  float* Oat = (float*)(ws + off);  off += (size_t)NB * TQ * CD * 4;
  float* kms = (float*)(ws + off);  off += (size_t)NB * TK * 4;
  float* qms = (float*)(ws + off);  off += (size_t)NB * TQ * 4;

  mask_kernel<<<NB * TQ, 256, 0, stream>>>(Q, qms);
  mask_kernel<<<NB * TK, 256, 0, stream>>>(K, kms);
  proj_kernel<<<dim3(CD / 128, (NB * TQ) / 128), 256, 0, stream>>>(Q, Wq, bq, Qb, Qlf);
  proj_kernel<<<dim3(CD / 128, (NB * TK) / 128), 256, 0, stream>>>(K, Wk, bk, Kb, nullptr);
  proj_kernel<<<dim3(CD / 128, (NB * TK) / 128), 256, 0, stream>>>(V, Wv, bv, Vb, nullptr);
  attn_kernel<<<NB * NHEADS * (TQ / 128), 256, 0, stream>>>(Qb, Kb, Vb, kms, qms, Oat);
  ln_kernel<<<NB * TQ, 256, 0, stream>>>(Oat, Qlf, lnw, lnb, (float*)d_out);
}
